// IFEBranch_31860067401853
// MI455X (gfx1250) — matrix operand order repaired and checked by reading, not compiled, not run on hardware
//
#include <hip/hip_runtime.h>
#include <hip/hip_bf16.h>

typedef __attribute__((ext_vector_type(16))) _Float16 v16h;
typedef __attribute__((ext_vector_type(8)))  _Float16 v8h;
typedef __attribute__((ext_vector_type(8)))  float    v8f;
typedef __attribute__((ext_vector_type(4)))  float    v4f;

#if defined(__has_builtin)
#if __has_builtin(__builtin_amdgcn_global_load_async_to_lds_b32)
#define HAVE_ASYNC_LDS 1
#endif
#endif
#ifndef HAVE_ASYNC_LDS
#define HAVE_ASYNC_LDS 0
#endif

static __device__ __forceinline__ int clampi(int v, int lo, int hi) {
    return v < lo ? lo : (v > hi ? hi : v);
}
static __device__ __forceinline__ float relu(float v) { return v > 0.f ? v : 0.f; }

// ---------------------------------------------------------------------------
// Kernel 1: rgb_uv_hist.  One block per image (16 blocks x 256 threads).
// LDS-binned scatter with ds_add_f32 atomics, tree reduction for norms,
// writes f16 histogram (B,3072) for the WMMA MLP.
// ---------------------------------------------------------------------------
__global__ void hist_kernel(const float* __restrict__ img, _Float16* __restrict__ hist16) {
    __shared__ float bins[3 * 1024];
    __shared__ float red[256];
    __shared__ float norms[3];
    const int n   = blockIdx.x;
    const int tid = threadIdx.x;

    for (int i = tid; i < 3 * 1024; i += 256) bins[i] = 0.f;
    __syncthreads();

    const float* im = img + (size_t)n * 3 * 512 * 512;
    const float eps   = 6.4f / 256.f;
    const float lo    = -3.2f - eps * 0.5f;
    const float hi    =  3.2f - eps * 0.5f;
    const float invbw = 32.f / (hi - lo);

    for (int p = tid; p < 1024; p += 256) {
        const int y = (p >> 5) * 16;   // arange(32)*512//32
        const int x = (p & 31) * 16;
        const float r = im[0 * 512 * 512 + y * 512 + x];
        const float g = im[1 * 512 * 512 + y * 512 + x];
        const float b = im[2 * 512 * 512 + y * 512 + x];
        const bool valid = (r > 0.f) && (g > 0.f) && (b > 0.f);
        const float sr = valid ? r : 1.f;
        const float sg = valid ? g : 1.f;
        const float sb = valid ? b : 1.f;
        const float Iy = sqrtf(sr * sr + sg * sg + sb * sb);
        const float lr = __logf(sr), lg = __logf(sg), lb = __logf(sb);
        const float Iu[3] = { lr - lb, lg - lb, lb - lg };
        const float Iv[3] = { lr - lg, lg - lr, lb - lr };
        #pragma unroll
        for (int c = 0; c < 3; ++c) {
            const float u = Iu[c], v = Iv[c];
            const int ub = clampi((int)floorf((u - lo) * invbw), 0, 31);
            const int vb = clampi((int)floorf((v - lo) * invbw), 0, 31);
            const bool inr = (u >= lo) && (u <= hi) && (v >= lo) && (v <= hi);
            const float w = (inr && valid) ? Iy : 0.f;
            atomicAdd(&bins[c * 1024 + ub * 32 + vb], w);
        }
    }
    __syncthreads();

    for (int c = 0; c < 3; ++c) {
        float s = 0.f;
        #pragma unroll
        for (int i = 0; i < 4; ++i) s += bins[c * 1024 + tid * 4 + i];
        red[tid] = s;
        __syncthreads();
        for (int off = 128; off > 0; off >>= 1) {
            if (tid < off) red[tid] += red[tid + off];
            __syncthreads();
        }
        if (tid == 0) norms[c] = red[0];
        __syncthreads();
    }

    for (int i = tid; i < 3072; i += 256) {
        const int c = i >> 10;
        float h = sqrtf(bins[i] / (norms[c] + 1e-6f));
        h = h < 0.f ? 0.f : (h > 1.f ? 1.f : h);
        hist16[(size_t)n * 3072 + i] = (_Float16)h;
    }
}

// ---------------------------------------------------------------------------
// Kernel 2: MLP layer via v_wmma_f32_16x16x32_f16.  M == 16 (batch) is one
// WMMA tile; one wave per 16-col tile.  A-frag loaded as two b128 chunks per
// lane (rows are 16B aligned: K multiple of 8).
// ---------------------------------------------------------------------------
template<bool OUT32>
__global__ void mlp_wmma_kernel(const _Float16* __restrict__ A,
                                const float* __restrict__ Bw,
                                const float* __restrict__ bias,
                                _Float16* __restrict__ out16,
                                float* __restrict__ out32,
                                int K, int N) {
    const int lane  = threadIdx.x;           // 32 threads = 1 wave
    const int ntile = blockIdx.x;
    const int m_a   = lane & 15;
    const int kb_a  = (lane >> 4) * 16;
    const int n_b   = ntile * 16 + (lane & 15);
    const int kb_b  = (lane >> 4) * 16;
    const _Float16* arow = A + (size_t)m_a * K + kb_a;

    v8f acc = {};
    for (int k0 = 0; k0 < K; k0 += 32) {
        const v8h alo = *(const v8h*)(arow + k0);        // K e=0..7
        const v8h ahi = *(const v8h*)(arow + k0 + 8);    // K e=8..15
        v16h a, b;
        #pragma unroll
        for (int e = 0; e < 8; ++e) { a[e] = alo[e]; a[e + 8] = ahi[e]; }
        #pragma unroll
        for (int e = 0; e < 16; ++e)
            b[e] = (_Float16)Bw[(size_t)(k0 + kb_b + e) * N + n_b];
        acc = __builtin_amdgcn_wmma_f32_16x16x32_f16(false, a, false, b,
                                                     (short)0, acc, false, false);
    }
    const float bn = bias[n_b];
    #pragma unroll
    for (int r = 0; r < 8; ++r) {
        const int m = r + 8 * (lane >> 4);
        const float v = relu(acc[r] + bn);
        if constexpr (OUT32) out32[(size_t)m * N + n_b] = v;
        else                 out16[(size_t)m * N + n_b] = (_Float16)v;
    }
}

// ---------------------------------------------------------------------------
// Kernel 3: Haar wavelet, f32 -> f16.  Pure bandwidth (50MB in / 25MB out).
// ---------------------------------------------------------------------------
__global__ void wavelet_kernel(const float* __restrict__ img, _Float16* __restrict__ x) {
    const int idx = blockIdx.x * 256 + threadIdx.x;   // 16*3*256*256
    if (idx >= 16 * 3 * 256 * 256) return;
    const int ox = idx & 255;
    int t = idx >> 8;
    const int oy = t & 255; t >>= 8;
    const int c = t % 3;
    const int n = t / 3;
    const float* im = img + (((size_t)n * 3 + c) * 512 + oy * 2) * 512 + ox * 2;
    const float a = im[0], b = im[1], cc = im[512], d = im[513];
    const size_t CH = 256 * 256;
    const size_t base = (size_t)n * 12 * CH + (size_t)oy * 256 + ox;
    x[base + (size_t)(c    ) * CH] = (_Float16)(( a + b + cc + d) * 0.5f);
    x[base + (size_t)(3 + c) * CH] = (_Float16)((-a + b - cc + d) * 0.5f);
    x[base + (size_t)(6 + c) * CH] = (_Float16)((-a - b + cc + d) * 0.5f);
    x[base + (size_t)(9 + c) * CH] = (_Float16)(( a - b - cc + d) * 0.5f);
}

// ---------------------------------------------------------------------------
// Kernel 4: 3x3 stride-2 pad-1 conv as implicit GEMM on the WMMA pipe.
// Tile = 16 output pixels (one row segment) x 16 output channels per wave.
// 8 waves/block share an LDS input patch (CIN x 3 rows x 36 cols f16; the 36
// cols start at an even x so interior dwords are 4B-aligned and can be staged
// with GLOBAL_LOAD_ASYNC_TO_LDS_B32, drained by s_wait_asynccnt).
// Weights: per-lane 16 contiguous f32 -> 4x b128 loads when K%32==0.
// MEAN_MODE fuses the spatial mean via global_atomic_add_f32.
// ---------------------------------------------------------------------------
template<int CIN, int COUT, int HIN, int WIN, bool MEAN_MODE>
__global__ void conv3x3s2_wmma_kernel(const _Float16* __restrict__ in,
                                      const float* __restrict__ wgt,
                                      const float* __restrict__ bias,
                                      _Float16* __restrict__ out,
                                      float* __restrict__ mean,
                                      float inv_count) {
    constexpr int HOUT   = HIN / 2;
    constexpr int WOUT   = WIN / 2;
    constexpr int K      = CIN * 9;
    constexpr int K32    = ((K + 31) / 32) * 32;
    constexpr int PATCHW = 36;                 // cols, even-aligned start
    constexpr int ICSTR  = 3 * PATCHW;         // 108 halfs per input channel

    __shared__ _Float16 patch[CIN * ICSTR];

    const int tid  = threadIdx.x;              // 256 = 8 waves
    const int wave = tid >> 5;
    const int lane = tid & 31;

    const int gp0  = blockIdx.x * 16;          // first output pixel of tile
    const int nimg = gp0 / (HOUT * WOUT);
    const int rem  = gp0 % (HOUT * WOUT);
    const int oy   = rem / WOUT;
    const int ox0  = rem % WOUT;               // multiple of 16

    const _Float16* inN = in + (size_t)nimg * CIN * HIN * WIN;
    const int iybase = 2 * oy - 1;
    const int ixbase = 2 * ox0 - 2;            // even => dword aligned

#if HAVE_ASYNC_LDS
    // Async copy interior dwords straight into LDS; masked scalar fallback on
    // the zero-padded boundary.  ASYNCcnt-tracked, no VGPR round trip.
    for (int idx = tid; idx < CIN * 3 * (PATCHW / 2); idx += 256) {
        const int ic = idx / (3 * (PATCHW / 2));
        const int r  = idx % (3 * (PATCHW / 2));
        const int ky = r / (PATCHW / 2);
        const int dw = r % (PATCHW / 2);
        const int iy  = iybase + ky;
        const int ix0 = ixbase + dw * 2;
        _Float16* lp = &patch[ic * ICSTR + ky * PATCHW + dw * 2];
        if (iy >= 0 && iy < HIN && ix0 >= 0 && ix0 + 1 < WIN) {
            const _Float16* gp = &inN[((size_t)ic * HIN + iy) * WIN + ix0];
            __builtin_amdgcn_global_load_async_to_lds_b32(
                (__attribute__((address_space(1))) int*)gp,
                (__attribute__((address_space(3))) int*)lp, 0, 0);
        } else {
            _Float16 v0 = (_Float16)0.f, v1 = (_Float16)0.f;
            if (iy >= 0 && iy < HIN) {
                if (ix0 >= 0 && ix0 < WIN)         v0 = inN[((size_t)ic * HIN + iy) * WIN + ix0];
                if (ix0 + 1 >= 0 && ix0 + 1 < WIN) v1 = inN[((size_t)ic * HIN + iy) * WIN + ix0 + 1];
            }
            lp[0] = v0; lp[1] = v1;
        }
    }
    asm volatile("s_wait_asynccnt 0x0" ::: "memory");
#else
    for (int idx = tid; idx < CIN * 3 * PATCHW; idx += 256) {
        const int ic = idx / ICSTR;
        const int r  = idx % ICSTR;
        const int ky = r / PATCHW;
        const int dx = r % PATCHW;
        const int iy = iybase + ky;
        const int ix = ixbase + dx;
        _Float16 v = (_Float16)0.f;
        if (iy >= 0 && iy < HIN && ix >= 0 && ix < WIN)
            v = inN[((size_t)ic * HIN + iy) * WIN + ix];
        patch[idx] = v;
    }
#endif
    __syncthreads();

    const int octile = blockIdx.y * 8 + wave;
    if (octile * 16 >= COUT) return;           // wave-uniform; no barriers below

    const int m_a  = lane & 15;
    const int kb_a = (lane >> 4) * 16;
    const int oc_b = octile * 16 + (lane & 15);
    const int kb_b = (lane >> 4) * 16;
    const float* wrow = wgt + (size_t)oc_b * K;   // w[oc][ic][ky][kx] contiguous

    v8f acc = {};
    for (int k0 = 0; k0 < K32; k0 += 32) {
        if (k0 + 32 < K)
            __builtin_prefetch(wrow + k0 + 32, 0, 1);   // global_prefetch_b8
        v16h a, b;
        #pragma unroll
        for (int e = 0; e < 16; ++e) {
            const int ka = k0 + kb_a + e;
            _Float16 av = (_Float16)0.f;
            if (ka < K) {
                const int ic = ka / 9;
                const int rr = ka % 9;
                const int ky = rr / 3;
                const int kx = rr % 3;
                av = patch[ic * ICSTR + ky * PATCHW + (m_a * 2 + kx + 1)];
            }
            a[e] = av;
        }
        if constexpr (K32 == K) {
            // 16 contiguous f32 weights per lane -> 4x global_load_b128
            const v4f* wv = (const v4f*)(wrow + k0 + kb_b);
            const v4f w0 = wv[0], w1 = wv[1], w2 = wv[2], w3 = wv[3];
            #pragma unroll
            for (int e = 0; e < 4; ++e) {
                b[e]      = (_Float16)w0[e];
                b[e + 4]  = (_Float16)w1[e];
                b[e + 8]  = (_Float16)w2[e];
                b[e + 12] = (_Float16)w3[e];
            }
        } else {
            #pragma unroll
            for (int e = 0; e < 16; ++e) {
                const int kb = k0 + kb_b + e;
                b[e] = (kb < K) ? (_Float16)wrow[kb] : (_Float16)0.f;
            }
        }
        acc = __builtin_amdgcn_wmma_f32_16x16x32_f16(false, a, false, b,
                                                     (short)0, acc, false, false);
    }

    const float bb = bias[oc_b];
    if (MEAN_MODE) {
        float s = 0.f;
        #pragma unroll
        for (int r = 0; r < 8; ++r) s += relu(acc[r] + bb);
        atomicAdd(&mean[(size_t)nimg * COUT + oc_b], s * inv_count);
    } else {
        // lane holds 8 consecutive pixels for its oc -> one b128 store
        v8h ov;
        #pragma unroll
        for (int r = 0; r < 8; ++r) ov[r] = (_Float16)relu(acc[r] + bb);
        _Float16* op = out + (((size_t)nimg * COUT + oc_b) * HOUT + oy) * WOUT
                           + ox0 + 8 * (lane >> 4);
        *(v8h*)op = ov;
    }
}

// ---------------------------------------------------------------------------
// Small helpers
// ---------------------------------------------------------------------------
__global__ void zero_kernel(float* __restrict__ p, int n) {
    const int i = blockIdx.x * 256 + threadIdx.x;
    if (i < n) p[i] = 0.f;
}

__global__ void final_kernel(const float* __restrict__ mean,
                             const float* __restrict__ hfeat,
                             float* __restrict__ out) {
    const int i = blockIdx.x * 256 + threadIdx.x;
    if (i < 16 * 256) out[i] = mean[i] * hfeat[i];
}

// ---------------------------------------------------------------------------
// Host launch
// ---------------------------------------------------------------------------
extern "C" void kernel_launch(void* const* d_in, const int* in_sizes, int n_in,
                              void* d_out, int out_size, void* d_ws, size_t ws_size,
                              hipStream_t stream) {
    const float* inp     = (const float*)d_in[0];
    const float* mlp_w1  = (const float*)d_in[1];
    const float* mlp_b1  = (const float*)d_in[2];
    const float* mlp_w2  = (const float*)d_in[3];
    const float* mlp_b2  = (const float*)d_in[4];
    const float* mlp_w3  = (const float*)d_in[5];
    const float* mlp_b3  = (const float*)d_in[6];
    const float* conv_w1 = (const float*)d_in[7];
    const float* conv_b1 = (const float*)d_in[8];
    const float* conv_w2 = (const float*)d_in[9];
    const float* conv_b2 = (const float*)d_in[10];
    const float* conv_w3 = (const float*)d_in[11];
    const float* conv_b3 = (const float*)d_in[12];
    float* out = (float*)d_out;

    // Workspace layout (all chunks 256B-aligned sizes)
    char* ws = (char*)d_ws;
    _Float16* hist16   = (_Float16*)ws; ws += (size_t)16 * 3072 * 2;           // 96 KB
    _Float16* m1       = (_Float16*)ws; ws += (size_t)16 * 1024 * 2;           // 32 KB
    _Float16* m2       = (_Float16*)ws; ws += (size_t)16 * 512 * 2;            // 16 KB
    float*    histfeat = (float*)ws;    ws += (size_t)16 * 256 * 4;            // 16 KB
    float*    meanbuf  = (float*)ws;    ws += (size_t)16 * 256 * 4;            // 16 KB
    _Float16* xwave    = (_Float16*)ws; ws += (size_t)16 * 12 * 256 * 256 * 2; // 24 MB
    _Float16* c1       = (_Float16*)ws; ws += (size_t)16 * 64 * 128 * 128 * 2; // 32 MB
    _Float16* c2       = (_Float16*)ws; ws += (size_t)16 * 128 * 64 * 64 * 2;  // 16 MB

    // Histogram branch
    hist_kernel<<<16, 256, 0, stream>>>(inp, hist16);
    mlp_wmma_kernel<false><<<1024 / 16, 32, 0, stream>>>(hist16, mlp_w1, mlp_b1, m1, nullptr, 3072, 1024);
    mlp_wmma_kernel<false><<<512 / 16,  32, 0, stream>>>(m1,     mlp_w2, mlp_b2, m2, nullptr, 1024, 512);
    mlp_wmma_kernel<true><<<256 / 16,   32, 0, stream>>>(m2,     mlp_w3, mlp_b3, nullptr, histfeat, 512, 256);

    // Wavelet branch
    wavelet_kernel<<<(16 * 3 * 256 * 256) / 256, 256, 0, stream>>>(inp, xwave);

    // conv1: (16,12,256,256) -> (16,64,128,128)
    {
        dim3 grid((16 * 128 * 128) / 16, 1);
        conv3x3s2_wmma_kernel<12, 64, 256, 256, false>
            <<<grid, 256, 0, stream>>>(xwave, conv_w1, conv_b1, c1, nullptr, 0.f);
    }
    // conv2: (16,64,128,128) -> (16,128,64,64)
    {
        dim3 grid((16 * 64 * 64) / 16, 1);
        conv3x3s2_wmma_kernel<64, 128, 128, 128, false>
            <<<grid, 256, 0, stream>>>(c1, conv_w2, conv_b2, c2, nullptr, 0.f);
    }
    // conv3 + fused spatial mean: (16,128,64,64) -> mean (16,256)
    zero_kernel<<<(16 * 256 + 255) / 256, 256, 0, stream>>>(meanbuf, 16 * 256);
    {
        dim3 grid((16 * 32 * 32) / 16, 2);
        conv3x3s2_wmma_kernel<128, 256, 64, 64, true>
            <<<grid, 256, 0, stream>>>(c2, conv_w3, conv_b3, nullptr, meanbuf, 1.f / 1024.f);
    }

    // Output: wave_feature * hist_feature
    final_kernel<<<(16 * 256 + 255) / 256, 256, 0, stream>>>(meanbuf, histfeat, out);
}
